// MSDeformAttn_21260088115210
// MI455X (gfx1250) — compile-verified, matrix operand order checked by reading, not run on hardware
//
#include <hip/hip_runtime.h>
#include <hip/hip_bf16.h>

typedef _Float16 v16h __attribute__((ext_vector_type(16)));
typedef _Float16 v8h  __attribute__((ext_vector_type(8)));
typedef float    v8f  __attribute__((ext_vector_type(8)));
typedef unsigned int u32x4 __attribute__((ext_vector_type(4)));
typedef int          i32x8 __attribute__((ext_vector_type(8)));
typedef int          i32x4 __attribute__((ext_vector_type(4)));

#define TT      5
#define HH      64
#define WW      96
#define HWX     (HH*WW)        // 6144
#define NTOK    (TT*HWX)       // 30720
#define CIN     256
#define KDIM    (CIN*9)        // 2304
#define KITERS  (KDIM/32)      // 72
#define NHEADS  8
#define NLEV    5
#define NPTS    4
#define DHEAD   32
#define BTOK    256            // tokens per block (8 waves x 2 fragments x 16)

// ---------------------------------------------------------------------------
// Weight repack: fp32 [Cout][Cin*9] -> f16 A-fragments.
// K is permuted (kh,kw)-major / ci-minor: kLin = (kh*3+kw)*256 + ci, so every
// 32-wide K slice has a single (kh,kw) and a contiguous ci range.
// Fragment layout matches the CDNA5 16-bit 16x32 A layout:
//   lanes 0-15: K {0..7, 16..23}; lanes 16-31: +8.
// aPacked index = ((coTile*KITERS + kk)*32 + lane)*16 + j
// ---------------------------------------------------------------------------
__global__ void pack_weights(const float* __restrict__ w, _Float16* __restrict__ out,
                             int coTiles) {
    int tid = blockIdx.x * 256 + threadIdx.x;
    int total = coTiles * KITERS * 32;
    if (tid >= total) return;
    int lane = tid & 31;
    int kk   = (tid >> 5) % KITERS;
    int ct   = (tid >> 5) / KITERS;
    int co   = ct * 16 + (lane & 15);
    int kAdd = (lane >= 16) ? 8 : 0;
    v16h v;
    #pragma unroll
    for (int j = 0; j < 16; ++j) {
        int kLin = kk * 32 + j + ((j >= 8) ? 8 : 0) + kAdd;
        int khkw = kLin >> 8;          // 0..8
        int ci   = kLin & 255;
        v[j] = (_Float16)w[(size_t)co * KDIM + (size_t)ci * 9 + khkw];
    }
    *(v16h*)(out + (size_t)tid * 16) = v;
}

// ---------------------------------------------------------------------------
// fp32 -> f16 activation pre-conversion (one pass; removes cvt from GEMM loop)
// ---------------------------------------------------------------------------
__global__ void cvt_f16(const float* __restrict__ in, _Float16* __restrict__ out, int n) {
    int i = blockIdx.x * 256 + threadIdx.x;
    if (i < n) out[i] = (_Float16)in[i];
}

// ---------------------------------------------------------------------------
// Implicit-GEMM 3x3 conv via v_wmma_f32_16x16x32_f16.
// Block: 256 threads = 8 waves. Tile: 256 tokens x 16 Cout (2 frags/wave).
// A panel (16 x 2304 f16 = 72KB) loaded once into LDS via the Tensor Data
// Mover; B slices double-buffered through LDS, one barrier per K step.
// ---------------------------------------------------------------------------
__global__ __launch_bounds__(256)
void conv_wmma(const _Float16* __restrict__ in, const _Float16* __restrict__ aPacked,
               const float* __restrict__ bias, float* __restrict__ out,
               float* __restrict__ vT,
               int inChanStride, int inTStride, int outCoStride, int outTStride) {
    __shared__ _Float16 ldsA[16 * KDIM];       // 73728 B, frag-packed A panel
    __shared__ _Float16 ldsB[2][BTOK * 40];    // [token][k], stride 40 halves

    const int tid    = threadIdx.x;
    const int wave   = tid >> 5;
    const int lane   = tid & 31;
    const int nBase  = blockIdx.x * BTOK;
    const int coTile = blockIdx.y;

    // ---- A panel -> LDS via TDM (fallback: cooperative b64 copy) ----
    const _Float16* gA = aPacked + (size_t)coTile * (16 * KDIM);
#if __has_builtin(__builtin_amdgcn_tensor_load_to_lds)
    if (wave == 0) {
        unsigned lds_off = (unsigned)(size_t)(&ldsA[0]);
        unsigned long long ga = (unsigned long long)(size_t)gA;
        u32x4 g0;
        g0[0] = 1u;                                           // count=1, no gather
        g0[1] = lds_off;                                      // lds_addr
        g0[2] = (unsigned)ga;                                 // global_addr[31:0]
        g0[3] = (unsigned)((ga >> 32) & 0x01FFFFFFu) | 0x80000000u; // ga[56:32] | type=2
        i32x8 g1;
        g1[0] = 3 << 16;                 // data_size = 8 bytes; mask/flags = 0
        g1[1] = (int)(9216u << 16);      // tensor_dim0 = 9216 (72KB / 8B)
        g1[2] = 0;                       // tensor_dim1 = 0 (1-D)
        g1[3] = (int)(9216u << 16);      // tile_dim0 = 9216
        g1[4] = 0;                       // tile_dim1 = 0 (unused)
        g1[5] = 9216;                    // tensor_dim0_stride
        g1[6] = 0;
        g1[7] = 0;
        i32x4 z4 = {0, 0, 0, 0};
        i32x8 z8 = {0, 0, 0, 0, 0, 0, 0, 0};
        __builtin_amdgcn_tensor_load_to_lds(g0, g1, z4, z4, z8, 0);
        __builtin_amdgcn_s_wait_tensorcnt(0);
    }
#else
    for (int i = tid; i < 16 * KDIM / 4; i += 256)
        ((unsigned long long*)ldsA)[i] = ((const unsigned long long*)gA)[i];
#endif

    // ---- per-thread staging geometry (token = tid; computed once) ----
    const int nTok = nBase + tid;
    const int t0   = nTok / HWX;
    const int hw0  = nTok - t0 * HWX;
    const int y0p  = hw0 / WW;
    const int x0p  = hw0 - y0p * WW;
    const size_t tOff = (size_t)t0 * inTStride;

    auto stage = [&](int kk, int bufIdx) {
        int khkw = kk >> 3;                  // (kh,kw) uniform within a K slice
        int kh = khkw / 3, kw = khkw - kh * 3;
        int iy = y0p + kh - 1, ix = x0p + kw - 1;
        bool valid = (iy >= 0) & (iy < HH) & (ix >= 0) & (ix < WW);
        int ciBase = (kk & 7) * 32;
        const _Float16* src = in + (size_t)ciBase * inChanStride + tOff
                                 + (size_t)iy * WW + ix;
        v16h v0 = {}, v1 = {};
        if (valid) {
            #pragma unroll
            for (int j = 0; j < 16; ++j) v0[j] = src[(size_t)j * inChanStride];
            #pragma unroll
            for (int j = 0; j < 16; ++j) v1[j] = src[(size_t)(16 + j) * inChanStride];
        }
        _Float16* dst = &ldsB[bufIdx][tid * 40];
        *(v8h*)(dst + 0)  = __builtin_shufflevector(v0, v0, 0,1,2,3,4,5,6,7);
        *(v8h*)(dst + 8)  = __builtin_shufflevector(v0, v0, 8,9,10,11,12,13,14,15);
        *(v8h*)(dst + 16) = __builtin_shufflevector(v1, v1, 0,1,2,3,4,5,6,7);
        *(v8h*)(dst + 24) = __builtin_shufflevector(v1, v1, 8,9,10,11,12,13,14,15);
    };

    v8f c0 = {0.f,0.f,0.f,0.f,0.f,0.f,0.f,0.f};
    v8f c1 = {0.f,0.f,0.f,0.f,0.f,0.f,0.f,0.f};

    stage(0, 0);
    for (int kk = 0; kk < KITERS; ++kk) {
        __syncthreads();                         // buf[kk&1] staged & prior reads done
        if (kk + 1 < KITERS) stage(kk + 1, (kk + 1) & 1);

        // A fragment from LDS panel (32B contiguous per lane)
        const v16h a = *(const v16h*)(&ldsA[(kk * 32 + lane) * 16]);

        // Two B fragments per wave
        const int col = lane & 15;
        const int kOff = (lane < 16) ? 0 : 8;
        const _Float16* bp0 = &ldsB[kk & 1][(wave * 32 + col) * 40 + kOff];
        const _Float16* bp1 = &ldsB[kk & 1][(wave * 32 + 16 + col) * 40 + kOff];
        v8h lo0 = *(const v8h*)bp0, hi0 = *(const v8h*)(bp0 + 16);
        v8h lo1 = *(const v8h*)bp1, hi1 = *(const v8h*)(bp1 + 16);
        v16h b0 = __builtin_shufflevector(lo0, hi0, 0,1,2,3,4,5,6,7,8,9,10,11,12,13,14,15);
        v16h b1 = __builtin_shufflevector(lo1, hi1, 0,1,2,3,4,5,6,7,8,9,10,11,12,13,14,15);

        c0 = __builtin_amdgcn_wmma_f32_16x16x32_f16(false, a, false, b0, (short)0, c0,
                                                    false, false);
        c1 = __builtin_amdgcn_wmma_f32_16x16x32_f16(false, a, false, b1, (short)0, c1,
                                                    false, false);
    }

    // ---- epilogue: C layout { M = r + 8*(lane>=16), N = lane&15 } ----
    const int coBase = coTile * 16;
    #pragma unroll
    for (int f = 0; f < 2; ++f) {
        const v8f& c = f ? c1 : c0;
        const int nCol = nBase + wave * 32 + f * 16 + (lane & 15);
        const int t    = nCol / HWX;
        const int hw   = nCol - t * HWX;
        #pragma unroll
        for (int r2 = 0; r2 < 8; ++r2) {
            int M  = r2 + ((lane >> 4) << 3);
            int co = coBase + M;
            float val = c[r2] + bias[co];
            out[(size_t)co * outCoStride + (size_t)t * outTStride + hw] = val;
            if (vT) vT[(size_t)nCol * CIN + co] = val;
        }
    }
}

// ---------------------------------------------------------------------------
// Flow composition: dst = a + warp(b, a)  (zero-padded bilinear)
// ---------------------------------------------------------------------------
__device__ inline float bilin1(const float* __restrict__ img, float px, float py) {
    int x0 = (int)floorf(px), y0 = (int)floorf(py);
    float fx = px - x0, fy = py - y0;
    float acc = 0.f;
    #pragma unroll
    for (int dy = 0; dy < 2; ++dy)
        #pragma unroll
        for (int dx = 0; dx < 2; ++dx) {
            int xi = x0 + dx, yi = y0 + dy;
            if (xi >= 0 && xi < WW && yi >= 0 && yi < HH) {
                float wgt = (dx ? fx : 1.f - fx) * (dy ? fy : 1.f - fy);
                acc += wgt * img[yi * WW + xi];
            }
        }
    return acc;
}

__global__ void flows_init(const float* __restrict__ ff, const float* __restrict__ fb,
                           float* __restrict__ flows) {
    int i = blockIdx.x * 256 + threadIdx.x;
    if (i >= 2 * HWX) return;
    flows[ 0 * 2 * HWX + i] = 0.f;                    // zero slot
    flows[ 1 * 2 * HWX + i] = ff[0 * 2 * HWX + i];    // f01
    flows[ 5 * 2 * HWX + i] = ff[1 * 2 * HWX + i];    // f12
    flows[ 8 * 2 * HWX + i] = ff[2 * 2 * HWX + i];    // f23
    flows[10 * 2 * HWX + i] = ff[3 * 2 * HWX + i];    // f34
    flows[11 * 2 * HWX + i] = fb[0 * 2 * HWX + i];    // b10
    flows[13 * 2 * HWX + i] = fb[1 * 2 * HWX + i];    // b21
    flows[16 * 2 * HWX + i] = fb[2 * 2 * HWX + i];    // b32
}

__global__ void flow_comp(const float* __restrict__ a, const float* __restrict__ b,
                          float* __restrict__ dst) {
    int hw = blockIdx.x * 256 + threadIdx.x;
    if (hw >= HWX) return;
    int y = hw / WW, x = hw - y * WW;
    float ax = a[hw], ay = a[HWX + hw];
    float px = (float)x + ax, py = (float)y + ay;
    dst[hw]       = ax + bilin1(b,       px, py);
    dst[HWX + hw] = ay + bilin1(b + HWX, px, py);
}

// ---------------------------------------------------------------------------
// Sampling: one wave per (token, head); lane = D_HEAD channel (wave32 native).
// Gathers 128B coalesced lines from the token-major value copy.
// Emits f16 directly (input layout for the final WMMA conv).
// ---------------------------------------------------------------------------
__global__ __launch_bounds__(256)
void samp_attn(const float* __restrict__ vT, const float* __restrict__ offB,
               const float* __restrict__ attnB, const float* __restrict__ ref,
               const float* __restrict__ flows, _Float16* __restrict__ outB) {
    const int gw   = blockIdx.x * 8 + (threadIdx.x >> 5);
    const int lane = threadIdx.x & 31;
    const int head = gw & 7;
    const int n    = gw >> 3;
    if (n >= NTOK) return;
    const int t  = n / HWX;
    const int hw = n - t * HWX;

    float aw[NLEV * NPTS];
    float m = -1e30f;
    #pragma unroll
    for (int i = 0; i < NLEV * NPTS; ++i) {
        aw[i] = attnB[(size_t)(head * 20 + i) * NTOK + n];
        m = fmaxf(m, aw[i]);
    }
    float s = 0.f;
    #pragma unroll
    for (int i = 0; i < NLEV * NPTS; ++i) { aw[i] = __expf(aw[i] - m); s += aw[i]; }
    const float inv = 1.f / s;

    // flow-add slot table (t=4 row duplicates t=3, matching the reference)
    const int addsTable[5][5] = {
        { 0,  1,  2, 3,  4},
        {11,  0,  5, 6,  7},
        {12, 13,  0, 8,  9},
        {14, 15, 16, 0, 10},
        {14, 15, 16, 0, 10}};

    float acc = 0.f;
    for (int lvl = 0; lvl < NLEV; ++lvl) {
        int slot = addsTable[t][lvl];
        float addx = flows[(size_t)slot * 2 * HWX + hw];
        float addy = flows[(size_t)slot * 2 * HWX + HWX + hw];
        float rx = ref[((size_t)n * NLEV + lvl) * 2 + 0];
        float ry = ref[((size_t)n * NLEV + lvl) * 2 + 1];
        #pragma unroll
        for (int p = 0; p < NPTS; ++p) {
            int co = head * 40 + lvl * 8 + p * 2;
            float ox = offB[(size_t)co * NTOK + n];
            float oy = offB[(size_t)(co + 1) * NTOK + n];
            float px = (rx + (ox + addx) * (1.f / WW)) * WW - 0.5f;
            float py = (ry + (oy + addy) * (1.f / HH)) * HH - 0.5f;
            int x0 = (int)floorf(px), y0 = (int)floorf(py);
            float fx = px - x0, fy = py - y0;
            float wgt = aw[lvl * 4 + p] * inv;
            float sval = 0.f;
            #pragma unroll
            for (int dy = 0; dy < 2; ++dy)
                #pragma unroll
                for (int dx = 0; dx < 2; ++dx) {
                    int xi = x0 + dx, yi = y0 + dy;
                    if (xi >= 0 && xi < WW && yi >= 0 && yi < HH) {
                        float w2 = (dx ? fx : 1.f - fx) * (dy ? fy : 1.f - fy);
                        sval += w2 * vT[((size_t)(lvl * HWX + yi * WW + xi)) * CIN
                                        + head * DHEAD + lane];
                    }
                }
            acc += wgt * sval;
        }
    }
    outB[(size_t)(head * DHEAD + lane) * NTOK + n] = (_Float16)acc;
}

// ---------------------------------------------------------------------------
extern "C" void kernel_launch(void* const* d_in, const int* in_sizes, int n_in,
                              void* d_out, int out_size, void* d_ws, size_t ws_size,
                              hipStream_t stream) {
    (void)in_sizes; (void)n_in; (void)out_size; (void)ws_size;
    const float* query  = (const float*)d_in[0];
    const float* inflat = (const float*)d_in[1];
    const float* refp   = (const float*)d_in[2];
    const float* ff     = (const float*)d_in[6];
    const float* fb     = (const float*)d_in[7];
    const float* w_off  = (const float*)d_in[8];
    const float* b_off  = (const float*)d_in[9];
    const float* w_attn = (const float*)d_in[10];
    const float* b_attn = (const float*)d_in[11];
    const float* w_val  = (const float*)d_in[12];
    const float* b_val  = (const float*)d_in[13];
    const float* w_out  = (const float*)d_in[14];
    const float* b_out  = (const float*)d_in[15];

    char* ws = (char*)d_ws;
    size_t cur = 0;
    auto alloc = [&](size_t bytes) -> void* {
        cur = (cur + 255) & ~(size_t)255;
        void* p = ws + cur;
        cur += bytes;
        return p;
    };

    float*     flows    = (float*)alloc((size_t)17 * 2 * HWX * sizeof(float));
    _Float16*  aVal     = (_Float16*)alloc((size_t)16 * KITERS * 512 * sizeof(_Float16));
    _Float16*  aOff     = (_Float16*)alloc((size_t)20 * KITERS * 512 * sizeof(_Float16));
    _Float16*  aAttn    = (_Float16*)alloc((size_t)10 * KITERS * 512 * sizeof(_Float16));
    _Float16*  aOut     = (_Float16*)alloc((size_t)16 * KITERS * 512 * sizeof(_Float16));
    _Float16*  qH       = (_Float16*)alloc((size_t)NTOK * CIN * sizeof(_Float16));
    _Float16*  ifH      = (_Float16*)alloc((size_t)NTOK * CIN * sizeof(_Float16));
    float*     valBuf   = (float*)alloc((size_t)256 * NTOK * sizeof(float));
    float*     valT     = (float*)alloc((size_t)NTOK * 256 * sizeof(float));
    float*     offBuf   = (float*)alloc((size_t)320 * NTOK * sizeof(float));
    float*     attnBuf  = (float*)alloc((size_t)160 * NTOK * sizeof(float));
    _Float16*  attnOutH = (_Float16*)alloc((size_t)256 * NTOK * sizeof(_Float16));

    // --- flows: init + dependent composition chain ---
    flows_init<<<(2 * HWX + 255) / 256, 256, 0, stream>>>(ff, fb, flows);
    const int gHW = (HWX + 255) / 256;
    auto F = [&](int s) { return flows + (size_t)s * 2 * HWX; };
    flow_comp<<<gHW, 256, 0, stream>>>(F(1),  F(5),  F(2));   // f02
    flow_comp<<<gHW, 256, 0, stream>>>(F(2),  F(8),  F(3));   // f03
    flow_comp<<<gHW, 256, 0, stream>>>(F(3),  F(10), F(4));   // f04
    flow_comp<<<gHW, 256, 0, stream>>>(F(5),  F(8),  F(6));   // f13
    flow_comp<<<gHW, 256, 0, stream>>>(F(6),  F(10), F(7));   // f14
    flow_comp<<<gHW, 256, 0, stream>>>(F(8),  F(10), F(9));   // f24
    flow_comp<<<gHW, 256, 0, stream>>>(F(16), F(13), F(15));  // b31
    flow_comp<<<gHW, 256, 0, stream>>>(F(15), F(11), F(14));  // b30
    flow_comp<<<gHW, 256, 0, stream>>>(F(13), F(11), F(12));  // b20

    // --- weight repacks + activation f16 conversion ---
    pack_weights<<<(16 * KITERS * 32 + 255) / 256, 256, 0, stream>>>(w_val,  aVal,  16);
    pack_weights<<<(20 * KITERS * 32 + 255) / 256, 256, 0, stream>>>(w_off,  aOff,  20);
    pack_weights<<<(10 * KITERS * 32 + 255) / 256, 256, 0, stream>>>(w_attn, aAttn, 10);
    pack_weights<<<(16 * KITERS * 32 + 255) / 256, 256, 0, stream>>>(w_out,  aOut,  16);
    cvt_f16<<<(NTOK * CIN + 255) / 256, 256, 0, stream>>>(query,  qH,  NTOK * CIN);
    cvt_f16<<<(NTOK * CIN + 255) / 256, 256, 0, stream>>>(inflat, ifH, NTOK * CIN);

    // --- WMMA convs ---
    const int nBlocks = NTOK / BTOK;   // 120
    conv_wmma<<<dim3(nBlocks, 16), 256, 0, stream>>>(
        ifH, aVal, b_val, valBuf, valT, HWX, CIN * HWX, NTOK, HWX);
    conv_wmma<<<dim3(nBlocks, 20), 256, 0, stream>>>(
        qH, aOff, b_off, offBuf, nullptr, HWX, CIN * HWX, NTOK, HWX);
    conv_wmma<<<dim3(nBlocks, 10), 256, 0, stream>>>(
        qH, aAttn, b_attn, attnBuf, nullptr, HWX, CIN * HWX, NTOK, HWX);

    // --- deformable sampling (writes f16 for the final conv) ---
    samp_attn<<<NTOK, 256, 0, stream>>>(valT, offBuf, attnBuf, refp, flows, attnOutH);

    // --- output conv straight into d_out in (t, c, h, w) layout ---
    conv_wmma<<<dim3(nBlocks, 16), 256, 0, stream>>>(
        attnOutH, aOut, b_out, (float*)d_out, nullptr, NTOK, HWX, HWX, CIN * HWX);
}
